// EdgeSelector_28948079575467
// MI455X (gfx1250) — compile-verified, matrix operand order checked by reading, not run on hardware
//
#include <hip/hip_runtime.h>
#include <hip/hip_bf16.h>

#define NN     48
#define NP1    49
#define DM     32
#define DMODEL 128
#define NHEAD  4
#define HD     32
#define SEQ    64   // padded sequence (49 -> 64)

typedef __attribute__((ext_vector_type(16))) _Float16 v16h;
typedef __attribute__((ext_vector_type(8)))  float    v8f;

// ---------------- WMMA fragment loaders (wave32, 16x16x32 f16) ----------------
// A-matrix 16x32 (row-major in memory, stride in halfs):
// lane<16 holds row (lane), K = {kc+0..7, kc+16..23}; lane>=16 holds K = {kc+8..15, kc+24..31}
__device__ inline v16h frag_rowmajor(const _Float16* base, int stride, int row0, int kc, int lane) {
  const _Float16* p = base + (size_t)(row0 + (lane & 15)) * stride + kc + ((lane >> 4) << 3);
  v16h f;
#pragma unroll
  for (int j = 0; j < 8; ++j) { f[j] = p[j]; f[8 + j] = p[16 + j]; }
  return f;
}
// B-matrix 32x16 where each output column's K-vector is contiguous in memory
__device__ inline v16h frag_colvec(const _Float16* colbase, int kc, int lane) {
  const _Float16* p = colbase + kc + ((lane >> 4) << 3);
  v16h f;
#pragma unroll
  for (int j = 0; j < 8; ++j) { f[j] = p[j]; f[8 + j] = p[16 + j]; }
  return f;
}
// B-matrix 32x16 where K dimension is strided in memory
__device__ inline v16h frag_strideK(const _Float16* base, int strideK, int kc, int lane) {
  int k0 = kc + ((lane >> 4) << 3);
  v16h f;
#pragma unroll
  for (int j = 0; j < 8; ++j) {
    f[j]     = base[(size_t)(k0 + j)      * strideK];
    f[8 + j] = base[(size_t)(k0 + 16 + j) * strideK];
  }
  return f;
}

__device__ inline float hash01(unsigned x) {
  x ^= x >> 17; x *= 0xED5AD4BBu; x ^= x >> 11; x *= 0xAC4C1B51u;
  x ^= x >> 15; x *= 0x31848BABu; x ^= x >> 14;
  return ((float)(x >> 8) + 0.5f) * (1.0f / 16777216.0f);
}

// ---------------- Prep: f16-transpose weights + pedestrian mask ----------------
__global__ void prep_kernel(const float* __restrict__ We, const float* __restrict__ Wq,
                            const float* __restrict__ am, _Float16* __restrict__ WeT,
                            _Float16* __restrict__ WqT, float* __restrict__ mp, int bsz) {
  int stride = gridDim.x * blockDim.x;
  int tid = blockIdx.x * blockDim.x + threadIdx.x;
  for (int i = tid; i < DMODEL * 96; i += stride) {           // WeT[n][k] = We[k][n]
    int n = i / 96, k = i % 96;
    WeT[i] = (_Float16)We[k * DMODEL + n];
  }
  for (int i = tid; i < 3 * DMODEL * DMODEL; i += stride) {   // WqT[n][k] = Wq[k][n]
    int n = i / DMODEL, k = i % DMODEL;
    WqT[i] = (_Float16)Wq[k * (3 * DMODEL) + n];
  }
  for (int i = tid; i < bsz * NN; i += stride) {              // mask_ped
    const float* r = am + (size_t)i * NN;
    float s = 0.f;
    for (int j = 0; j < NN; ++j) s += r[j];
    mp[i] = (s > 0.f) ? 1.f : 0.f;
  }
}

// ---------------- Fused per-(b,target) kernel ----------------
// LDS budget (152,832 B): R1 32KB = feat(f16 64x96) -> Eh(f16 64x128) -> probs(f16 4x64x64)
//                         R2 48KB = qkv f16 64x384
//                         R3 64KB = scores f32 4x64x64 -> A2 f32 4x64x32
//                         + 5.25KB constants/stats
__global__ __launch_bounds__(256) void edge_main(
    const float* __restrict__ x, const float* __restrict__ A, const float* __restrict__ am,
    const float* __restrict__ bE, const float* __restrict__ gamma, const float* __restrict__ beta,
    const float* __restrict__ bQ, const float* __restrict__ W1, const float* __restrict__ b1,
    const float* __restrict__ W2, const float* __restrict__ b2,
    const _Float16* __restrict__ WeT, const _Float16* __restrict__ WqT,
    const float* __restrict__ mp, float* __restrict__ outEm, float* __restrict__ outSm) {
  __shared__ __align__(16) char sm[152832];
  _Float16* sFeat = (_Float16*)sm;                       // 64x96  (12,288 B)
  _Float16* sEh   = (_Float16*)sm;                       // 64x128 (16,384 B), after GEMM1
  _Float16* sP    = (_Float16*)sm;                       // 4x64x64 (32,768 B), after GEMM2
  _Float16* sQKV  = (_Float16*)(sm + 32768);             // 64x384 (49,152 B)
  float*    sS    = (float*)(sm + 32768 + 49152);        // 4x64x64 f32 (65,536 B)
  float*    sA2   = (float*)(sm + 32768 + 49152);        // 4x64x32 f32, after softmax
  float*    cst   = (float*)(sm + 32768 + 49152 + 65536);
  float* sBE  = cst;          // 128
  float* sBQ  = cst + 128;    // 384
  float* sG   = cst + 512;    // 128
  float* sBt  = cst + 640;    // 128
  float* sM   = cst + 768;    // 64
  float* sMu  = cst + 832;    // 64
  float* sInv = cst + 896;    // 64
  float* sRS  = cst + 960;    // 64
  float* sRS2 = cst + 1024;   // 64
  float* sSc  = cst + 1088;   // 256

  const int t   = blockIdx.x;   // target node
  const int b   = blockIdx.y;   // batch
  const int tid = threadIdx.x;
  const int lane = tid & 31, wv = tid >> 5;

  // ---- preload constants / mask ----
  for (int i = tid; i < DMODEL; i += 256) { sBE[i] = bE[i]; sG[i] = gamma[i]; sBt[i] = beta[i]; }
  for (int i = tid; i < 3 * DMODEL; i += 256) sBQ[i] = bQ[i];
  if (tid < 64) {
    sM[tid]  = (tid < NN) ? am[((size_t)(b * NN + t)) * NN + tid] : ((tid == NN) ? 1.f : 0.f);
    sRS[tid] = 0.f; sRS2[tid] = 0.f;
  }
  // ---- build augmented edge features (64x96, f16) ----
  const float mpt = mp[b * NN + t];
  for (int i = tid; i < SEQ * 96; i += 256) {
    int j = i / 96, c = i % 96;
    float v = 0.f;
    if (j <= NN) {
      if (c < DM) {
        if (j < NN) v = x[(size_t)(b * NN + j) * DM + c] * mp[b * NN + j];
      } else if (c < 2 * DM) {
        v = x[(size_t)(b * NN + t) * DM + (c - DM)] * mpt;
      } else {
        if (j < NN) v = A[(((size_t)(b * NN + j)) * NN + t) * DM + (c - 2 * DM)]
                        * am[((size_t)(b * NN + t)) * NN + j];
      }
    }
    sFeat[i] = (_Float16)v;
  }
  __syncthreads();

  // ---- GEMM1: E(64x128) = feat(64x96) @ W_embed + bias, then LN -> Eh f16 ----
  {
    const int n0  = wv * 16;
    const int col = n0 + (lane & 15);
    v8f zero = {};
    v8f acc[4] = { zero, zero, zero, zero };
    const _Float16* colbase = WeT + (size_t)col * 96;
    for (int kc = 0; kc < 96; kc += 32) {
      v16h bf = frag_colvec(colbase, kc, lane);
#pragma unroll
      for (int mt = 0; mt < 4; ++mt) {
        v16h af = frag_rowmajor(sFeat, 96, mt * 16, kc, lane);
        acc[mt] = __builtin_amdgcn_wmma_f32_16x16x32_f16(false, af, false, bf, (short)0,
                                                         acc[mt], false, false);
      }
    }
    const float bias = sBE[col];
#pragma unroll
    for (int mt = 0; mt < 4; ++mt)
#pragma unroll
      for (int r = 0; r < 8; ++r) {
        float e = acc[mt][r] + bias;
        acc[mt][r] = e;
        int row = mt * 16 + r + ((lane >> 4) << 3);
        atomicAdd(&sRS[row], e);
        atomicAdd(&sRS2[row], e * e);
      }
    __syncthreads();
    if (tid < 64) {
      float mu  = sRS[tid] * (1.f / 128.f);
      float var = sRS2[tid] * (1.f / 128.f) - mu * mu;
      sMu[tid]  = mu;
      sInv[tid] = rsqrtf(var + 1e-5f);
    }
    __syncthreads();
    const float g = sG[col], be = sBt[col];
#pragma unroll
    for (int mt = 0; mt < 4; ++mt)
#pragma unroll
      for (int r = 0; r < 8; ++r) {
        int row = mt * 16 + r + ((lane >> 4) << 3);
        float v = (acc[mt][r] - sMu[row]) * sInv[row] * g + be;
        sEh[row * DMODEL + col] = (_Float16)v;
      }
  }
  __syncthreads();

  // ---- GEMM2: qkv(64x384) = Eh(64x128) @ W_qkv + bias (f16 out) ----
  for (int ntl = 0; ntl < 3; ++ntl) {
    const int n0  = (wv * 3 + ntl) * 16;
    const int col = n0 + (lane & 15);
    const _Float16* colbase = WqT + (size_t)col * DMODEL;
    v8f zero = {};
    v8f acc[4] = { zero, zero, zero, zero };
    for (int kc = 0; kc < DMODEL; kc += 32) {
      v16h bf = frag_colvec(colbase, kc, lane);
#pragma unroll
      for (int mt = 0; mt < 4; ++mt) {
        v16h af = frag_rowmajor(sEh, DMODEL, mt * 16, kc, lane);
        acc[mt] = __builtin_amdgcn_wmma_f32_16x16x32_f16(false, af, false, bf, (short)0,
                                                         acc[mt], false, false);
      }
    }
    const float bias = sBQ[col];
#pragma unroll
    for (int mt = 0; mt < 4; ++mt)
#pragma unroll
      for (int r = 0; r < 8; ++r) {
        int row = mt * 16 + r + ((lane >> 4) << 3);
        sQKV[(size_t)row * 384 + col] = (_Float16)(acc[mt][r] + bias);
      }
  }
  __syncthreads();

  // ---- attention scores per head: S = (q/sqrt(hd)) @ k^T, masked ----
  {
    const int h = wv >> 1;
    const float scale = 0.17677669529663687f;  // 1/sqrt(32)
    for (int ntl = 0; ntl < 2; ++ntl) {
      const int nt = ((wv & 1) << 1) + ntl;
      const int j0 = nt * 16;
      const int jcol = j0 + (lane & 15);
      const _Float16* kcol = sQKV + (size_t)jcol * 384 + DMODEL + h * HD;  // k[j, d] contiguous in d
      v16h bf = frag_colvec(kcol, 0, lane);
      const float mj = sM[jcol];
#pragma unroll
      for (int mt = 0; mt < 4; ++mt) {
        v16h af = frag_rowmajor(sQKV + h * HD, 384, mt * 16, 0, lane);    // q[i, d]
        v8f acc = {};
        acc = __builtin_amdgcn_wmma_f32_16x16x32_f16(false, af, false, bf, (short)0,
                                                     acc, false, false);
#pragma unroll
        for (int r = 0; r < 8; ++r) {
          int row = mt * 16 + r + ((lane >> 4) << 3);
          bool valid = (row < NP1) && (jcol < NP1) && (sM[row] > 0.5f) && (mj > 0.5f);
          sS[((size_t)h * SEQ + row) * SEQ + jcol] = valid ? acc[r] * scale : -1e9f;
        }
      }
    }
  }
  __syncthreads();

  // ---- row softmax: scores f32 (R3) -> probs f16 (R1) ----
  {
    const int h = tid >> 6, i = tid & 63;
    const float* row = sS + ((size_t)h * SEQ + i) * SEQ;
    float mx = -3.4e38f;
    for (int j = 0; j < SEQ; ++j) mx = fmaxf(mx, row[j]);
    float sum = 0.f;
    for (int j = 0; j < SEQ; ++j) sum += __expf(row[j] - mx);
    const float inv = 1.f / sum;
    _Float16* prow = sP + ((size_t)h * SEQ + i) * SEQ;
    for (int j = 0; j < SEQ; ++j) prow[j] = (_Float16)(__expf(row[j] - mx) * inv);
  }
  __syncthreads();

  // ---- A2 = probs(64x64) @ v(64x32) per head (f32 out, overlays scores) ----
  {
    const int h = wv >> 1, nt = wv & 1;
    const int d0 = nt * 16;
    v8f zero = {};
    v8f acc[4] = { zero, zero, zero, zero };
    const _Float16* vbase = sQKV + 2 * DMODEL + h * HD + d0 + (lane & 15);  // v[j, d], K=j strided
    for (int kc = 0; kc < SEQ; kc += 32) {
      v16h bf = frag_strideK(vbase, 384, kc, lane);
#pragma unroll
      for (int mt = 0; mt < 4; ++mt) {
        v16h af = frag_rowmajor(sP + (size_t)h * SEQ * SEQ, SEQ, mt * 16, kc, lane);
        acc[mt] = __builtin_amdgcn_wmma_f32_16x16x32_f16(false, af, false, bf, (short)0,
                                                         acc[mt], false, false);
      }
    }
#pragma unroll
    for (int mt = 0; mt < 4; ++mt)
#pragma unroll
      for (int r = 0; r < 8; ++r) {
        int row = mt * 16 + r + ((lane >> 4) << 3);
        sA2[((size_t)h * SEQ + row) * HD + d0 + (lane & 15)] = acc[mt][r];
      }
  }
  __syncthreads();

  // ---- gating MLP: s = relu(A2 @ W1 + b1) @ W2 + b2 (one row per thread) ----
  {
    const int h = tid >> 6, i = tid & 63;
    const float* arow = sA2 + ((size_t)h * SEQ + i) * HD;
    float a[HD];
#pragma unroll
    for (int kk = 0; kk < HD; ++kk) a[kk] = arow[kk];
    float s = b2[0];
    for (int j = 0; j < HD; ++j) {
      float hh = b1[j];
#pragma unroll
      for (int kk = 0; kk < HD; ++kk) hh += a[kk] * W1[kk * HD + j];
      s += fmaxf(hh, 0.f) * W2[j];
    }
    sSc[h * SEQ + i] = s;
  }
  __syncthreads();

  // ---- em softmax*mask + renorm, gumbel-softmax sample (one head per thread) ----
  if (tid < NHEAD) {
    const int h = tid;
    float* s = sSc + h * SEQ;
    const size_t obase = (((size_t)(b * NN + t)) * NHEAD + h) * NP1;
    float mx = -3.4e38f;
    for (int i = 0; i < NP1; ++i) mx = fmaxf(mx, s[i]);
    float sum = 0.f;
    for (int i = 0; i < NP1; ++i) { float e = __expf(s[i] - mx); sum += e; s[i] = e; }
    float sum2 = 0.f;
    for (int i = 0; i < NP1; ++i) { float e = (s[i] / sum) * sM[i]; sum2 += e; s[i] = e; }
    const float inv = 1.f / (sum2 + 1e-10f);
    float zmax = -3.4e38f;
    for (int i = 0; i < NP1; ++i) {
      float em = s[i] * inv;
      outEm[obase + i] = em;
      float u = hash01((unsigned)(obase + i) * 2654435761u + 0x9E3779B9u);
      float g = -__logf(-__logf(u));                  // Gumbel(0,1)
      float z = __logf(em + 1e-10f) + g;              // TAU = 1
      s[i] = z;
      zmax = fmaxf(zmax, z);
    }
    float zs = 0.f;
    for (int i = 0; i < NP1; ++i) { float ez = __expf(s[i] - zmax); zs += ez; s[i] = ez; }
    const float zinv = 1.f / zs;
    for (int i = 0; i < NP1; ++i) outSm[obase + i] = s[i] * zinv;
  }
}

extern "C" void kernel_launch(void* const* d_in, const int* in_sizes, int n_in,
                              void* d_out, int out_size, void* d_ws, size_t ws_size,
                              hipStream_t stream) {
  const float* x    = (const float*)d_in[0];
  const float* A    = (const float*)d_in[1];
  const float* am   = (const float*)d_in[2];
  const float* We   = (const float*)d_in[3];
  const float* bE   = (const float*)d_in[4];
  const float* gm   = (const float*)d_in[5];
  const float* bt   = (const float*)d_in[6];
  const float* Wq   = (const float*)d_in[7];
  const float* bQ   = (const float*)d_in[8];
  const float* W1   = (const float*)d_in[9];
  const float* b1   = (const float*)d_in[10];
  const float* W2   = (const float*)d_in[11];
  const float* b2   = (const float*)d_in[12];

  const int bsz = in_sizes[0] / (NN * DM);   // 128

  char* ws = (char*)d_ws;
  _Float16* WeT = (_Float16*)ws;                       // 128*96 halfs  = 24,576 B
  _Float16* WqT = (_Float16*)(ws + 24576);             // 384*128 halfs = 98,304 B
  float*    mp  = (float*)(ws + 24576 + 98304);        // bsz*48 f32

  float* outEm = (float*)d_out;
  float* outSm = outEm + (size_t)bsz * NN * NHEAD * NP1;

  prep_kernel<<<96, 256, 0, stream>>>(We, Wq, am, WeT, WqT, mp, bsz);

  dim3 grid(NN, bsz);
  edge_main<<<grid, 256, 0, stream>>>(x, A, am, bE, gm, bt, bQ, W1, b1, W2, b2,
                                      WeT, WqT, mp, outEm, outSm);
}
